// QuantumNatModel_65481071399585
// MI455X (gfx1250) — matrix operand order repaired and checked by reading, not compiled, not run on hardware
//
#include <hip/hip_runtime.h>
#include <math.h>

typedef __attribute__((ext_vector_type(16))) _Float16 v16h;
typedef __attribute__((ext_vector_type(8)))  float    v8f;

#define EPSB 1e-5f

// Async global->LDS 128-bit copy (CDNA5 GLOBAL_LOAD_ASYNC_TO_LDS_B128,
// tracked by ASYNCcnt). The LDS destination offset is derived from the real
// shared-array address so the pointer escapes into the asm and the compiler
// keeps both the LDS allocation and the later ds_load reads.
__device__ __forceinline__ void async_g2l_b128(void* lds_ptr, const void* gaddr) {
    unsigned loff = (unsigned)(unsigned long long)lds_ptr;   // addr[31:0] = LDS offset
    unsigned long long ga = (unsigned long long)gaddr;
    asm volatile("global_load_async_to_lds_b128 %0, %1, off"
                 :: "v"(loff), "v"(ga) : "memory");
}
__device__ __forceinline__ void wait_async0() {
    asm volatile("s_wait_asynccnt 0x0" ::: "memory");
}

// ---------------- stats layout (floats at ws base) ----------------
// [0..7]   bn1 sum      [8..15]  bn1 sumsq
// [16..31] bn2 sum      [32..47] bn2 sumsq
// [48..51] bn3 sum      [52..55] bn3 sumsq
__global__ void qn_init_stats(float* stats) {
    if (threadIdx.x < 64) stats[threadIdx.x] = 0.f;
}

// ---------------- conv1: 1->8ch, 3x3 SAME on 32x32, WMMA implicit GEMM ----
// Block = 4 waves = 2 output rows (y0, y0+1) x 2 segments of 16 px.
// Input halo rows y0-1..y0+2 staged in LDS via async global->LDS b128.
__global__ void qn_conv1_wmma(const float* __restrict__ x,
                              const float* __restrict__ w1,
                              const float* __restrict__ b1,
                              _Float16* __restrict__ h1) {
    __shared__ float lds1[4 * 32];           // 4 halo rows x 32 cols
    int lane = threadIdx.x & 31;
    int wave = threadIdx.x >> 5;
    int b    = blockIdx.x >> 4;              // 65536 blocks, 16 per image
    int y0   = (blockIdx.x & 15) * 2;
    int y    = y0 + (wave >> 1);
    int xbase = (wave & 1) << 4;
    int n    = lane & 15;
    int hl   = lane >> 4;

    const float* xb = x + (size_t)b * 1024;
    if (threadIdx.x < 32) {                  // wave 0: stage 4 rows x 32 floats
        int row = threadIdx.x >> 3;          // 0..3 -> y0-1+row (clamped)
        int chunk = threadIdx.x & 7;         // 8 x b128 per row
        int gy = y0 - 1 + row;
        gy = gy < 0 ? 0 : (gy > 31 ? 31 : gy);
        async_g2l_b128(&lds1[row * 32 + chunk * 4],
                       xb + gy * 32 + chunk * 4);
    }
    // B fragment: weights, K = j + 16*hl, N = n
    v16h bf;
#pragma unroll
    for (int j = 0; j < 16; ++j) {
        int k = j + 16 * hl;
        float v = (n < 8 && k < 9) ? w1[n * 9 + k] : 0.f;
        bf[j] = (_Float16)v;
    }
    wait_async0();
    __syncthreads();
    // A fragment from LDS: M = lane&15 -> pixel x, K = j + 16*hl (same as B)
    int px = xbase + (lane & 15);
    v16h af;
#pragma unroll
    for (int j = 0; j < 16; ++j) {
        int k = j + 16 * hl;
        float v = 0.f;
        if (k < 9) {
            int yy = y + k / 3 - 1;          // in [y0-1, y0+2] always
            int xx = px + k % 3 - 1;
            if (yy >= 0 && yy < 32 && xx >= 0 && xx < 32)
                v = lds1[(yy - (y0 - 1)) * 32 + xx];
        }
        af[j] = (_Float16)v;
    }
    v8f c = {};
    c = __builtin_amdgcn_wmma_f32_16x16x32_f16(false, af, false, bf,
                                               (short)0, c, false, false);
    if (n < 8) {
        float bias = b1[n];
        _Float16* out = h1 + ((size_t)(b * 8 + n) * 32 + y) * 32;
#pragma unroll
        for (int i = 0; i < 8; ++i)
            out[xbase + i + 8 * hl] = (_Float16)(c[i] + bias);
    }
}

// ---------------- per-(b,c) plane reduction for BN1 stats -----------------
__global__ void qn_stats1(const _Float16* __restrict__ h1, float* stats) {
    int bc = blockIdx.x;                     // 32768 = 4096*8
    int c  = bc & 7;
    const _Float16* p = h1 + (size_t)bc * 1024;
    float s = 0.f, sq = 0.f;
    for (int i = threadIdx.x; i < 1024; i += 256) {
        float v = (float)p[i];
        s += v; sq += v * v;
    }
    __shared__ float ls[256], lq[256];
    ls[threadIdx.x] = s; lq[threadIdx.x] = sq;
    __syncthreads();
    for (int off = 128; off > 0; off >>= 1) {
        if (threadIdx.x < off) {
            ls[threadIdx.x] += ls[threadIdx.x + off];
            lq[threadIdx.x] += lq[threadIdx.x + off];
        }
        __syncthreads();
    }
    if (threadIdx.x == 0) {
        atomicAdd(&stats[c], ls[0]);
        atomicAdd(&stats[8 + c], lq[0]);
    }
}

// ---------------- BN1 + ReLU + 2x2 maxpool -> h1p [4096,8,16,16] f16 ------
__global__ void qn_pool1(const _Float16* __restrict__ h1,
                         const float* __restrict__ stats,
                         const float* __restrict__ g1,
                         const float* __restrict__ bb1,
                         _Float16* __restrict__ h1p) {
    int o = blockIdx.x * 256 + threadIdx.x;  // 8,388,608
    int b = o >> 11;
    int c = (o >> 8) & 7;
    int y = (o >> 4) & 15;
    int xx = o & 15;
    const float invN = 1.f / 4194304.f;
    float mean = stats[c] * invN;
    float var  = stats[8 + c] * invN - mean * mean;
    float sc   = rsqrtf(var + EPSB) * g1[c];
    float be   = bb1[c];
    const _Float16* p = h1 + ((size_t)(b * 8 + c) * 32 + 2 * y) * 32 + 2 * xx;
    float m = -1e30f;
#pragma unroll
    for (int dy = 0; dy < 2; ++dy)
#pragma unroll
        for (int dx = 0; dx < 2; ++dx) {
            float v = fmaxf(((float)p[dy * 32 + dx] - mean) * sc + be, 0.f);
            m = fmaxf(m, v);
        }
    h1p[o] = (_Float16)m;
}

// ---------------- conv2: 8->16ch, 3x3 SAME on 16x16, WMMA K=72->96 --------
// Block = 4 waves = 4 output rows of one image. Whole 8x16x16 f16 input
// image (4 KB) staged into LDS via async global->LDS b128.
__global__ void qn_conv2_wmma(const _Float16* __restrict__ h1p,
                              const float* __restrict__ w2,
                              const float* __restrict__ b2,
                              _Float16* __restrict__ h2) {
    __shared__ _Float16 tile[2048];          // 8ch x 16 x 16
    int lane = threadIdx.x & 31;
    int wave = threadIdx.x >> 5;
    int b    = blockIdx.x >> 2;              // 16384 blocks, 4 per image
    int y    = (blockIdx.x & 3) * 4 + wave;
    int n    = lane & 15;
    int hl   = lane >> 4;
    int px   = lane & 15;                    // A-matrix M -> output x

    const _Float16* inb = h1p + (size_t)b * 2048;
#pragma unroll
    for (int k = 0; k < 2; ++k) {            // 256 x 16B chunks, 128 threads
        int cidx = threadIdx.x + 128 * k;
        async_g2l_b128(&tile[cidx * 8], inb + cidx * 8);
    }
    wait_async0();
    __syncthreads();

    v8f c = {};
#pragma unroll
    for (int s = 0; s < 3; ++s) {
        v16h af, bf;
#pragma unroll
        for (int j = 0; j < 16; ++j) {
            int kb = 32 * s + j + 16 * hl;
            float wv = (kb < 72) ? w2[n * 72 + kb] : 0.f;
            bf[j] = (_Float16)wv;
            int ka = 32 * s + j + 16 * hl;
            float av = 0.f;
            if (ka < 72) {
                int ic = ka / 9, tap = ka % 9;
                int yy = y + tap / 3 - 1;
                int xx = px + tap % 3 - 1;
                if (yy >= 0 && yy < 16 && xx >= 0 && xx < 16)
                    av = (float)tile[(ic * 16 + yy) * 16 + xx];
            }
            af[j] = (_Float16)av;
        }
        c = __builtin_amdgcn_wmma_f32_16x16x32_f16(false, af, false, bf,
                                                   (short)0, c, false, false);
    }
    float bias = b2[n];
    _Float16* out = h2 + ((size_t)(b * 16 + n) * 16 + y) * 16;
#pragma unroll
    for (int i = 0; i < 8; ++i)
        out[i + 8 * hl] = (_Float16)(c[i] + bias);
}

// ---------------- BN2 stats: block per (b,c), plane = 256 -----------------
__global__ void qn_stats2(const _Float16* __restrict__ h2, float* stats) {
    int bc = blockIdx.x;                     // 65536 = 4096*16
    int c  = bc & 15;
    float v = (float)h2[(size_t)bc * 256 + threadIdx.x];
    __shared__ float ls[256], lq[256];
    ls[threadIdx.x] = v; lq[threadIdx.x] = v * v;
    __syncthreads();
    for (int off = 128; off > 0; off >>= 1) {
        if (threadIdx.x < off) {
            ls[threadIdx.x] += ls[threadIdx.x + off];
            lq[threadIdx.x] += lq[threadIdx.x + off];
        }
        __syncthreads();
    }
    if (threadIdx.x == 0) {
        atomicAdd(&stats[16 + c], ls[0]);
        atomicAdd(&stats[32 + c], lq[0]);
    }
}

// ---------------- BN2 + ReLU + pool fused with per-sample mean ------------
__global__ void qn_encode(const _Float16* __restrict__ h2,
                          const float* __restrict__ stats,
                          const float* __restrict__ g2,
                          const float* __restrict__ bb2,
                          float* __restrict__ encode) {
    int b = blockIdx.x;                      // 4096
    const float invN = 1.f / 1048576.f;
    const _Float16* base = h2 + (size_t)b * 16 * 256;
    float s = 0.f;
#pragma unroll
    for (int k = 0; k < 4; ++k) {
        int po = threadIdx.x + 256 * k;      // 1024 pooled outputs
        int c  = po >> 6;
        int rr = po & 63;
        int py = rr >> 3, pxx = rr & 7;
        float mean = stats[16 + c] * invN;
        float var  = stats[32 + c] * invN - mean * mean;
        float sc   = rsqrtf(var + EPSB) * g2[c];
        float be   = bb2[c];
        const _Float16* p = base + (c * 16 + 2 * py) * 16 + 2 * pxx;
        float m = -1e30f;
#pragma unroll
        for (int dy = 0; dy < 2; ++dy)
#pragma unroll
            for (int dx = 0; dx < 2; ++dx) {
                float v = fmaxf(((float)p[dy * 16 + dx] - mean) * sc + be, 0.f);
                m = fmaxf(m, v);
            }
        s += m;
    }
    __shared__ float ls[256];
    ls[threadIdx.x] = s;
    __syncthreads();
    for (int off = 128; off > 0; off >>= 1) {
        if (threadIdx.x < off) ls[threadIdx.x] += ls[threadIdx.x + off];
        __syncthreads();
    }
    if (threadIdx.x == 0) encode[b] = ls[0] * (1.f / 1024.f);
}

// ---------------- 4-qubit circuit + FC + BN3 stat accumulation ------------
__global__ void qn_quantum_fc(const float* __restrict__ encode,
                              const float* __restrict__ rz,
                              const float* __restrict__ fcw,
                              const float* __restrict__ fcb,
                              float* __restrict__ out_pre,
                              float* stats) {
    int b = blockIdx.x * 256 + threadIdx.x;  // 4096
    float th = encode[b] * 0.5f;
    float c0 = cosf(th), s0 = sinf(th);
    float re[16], im[16];
#pragma unroll
    for (int i = 0; i < 16; ++i) {
        float a = 1.f;
#pragma unroll
        for (int q = 0; q < 4; ++q) a *= ((i >> (3 - q)) & 1) ? s0 : c0;
        re[i] = a; im[i] = 0.f;
    }
#pragma unroll
    for (int l = 0; l < 2; ++l) {
        // CNOT chain 0->1, 1->2, 2->3 (qubit 0 = MSB)
#pragma unroll
        for (int cq = 0; cq < 3; ++cq) {
            int cb = 1 << (3 - cq), tb = 1 << (2 - cq);
#pragma unroll
            for (int i = 0; i < 16; ++i) {
                if ((i & cb) && !(i & tb)) {
                    int j2 = i | tb;
                    float t;
                    t = re[i]; re[i] = re[j2]; re[j2] = t;
                    t = im[i]; im[i] = im[j2]; im[j2] = t;
                }
            }
        }
        // RZ per qubit: bit0 -> e^{-i phi/2}, bit1 -> e^{+i phi/2}
#pragma unroll
        for (int q = 0; q < 4; ++q) {
            float ph = rz[l * 4 + q] * 0.5f;
            float cp = cosf(ph), sp = sinf(ph);
#pragma unroll
            for (int i = 0; i < 16; ++i) {
                float sg = ((i >> (3 - q)) & 1) ? sp : -sp;
                float nr = re[i] * cp - im[i] * sg;
                float ni = re[i] * sg + im[i] * cp;
                re[i] = nr; im[i] = ni;
            }
        }
    }
    float ev[4] = {0.f, 0.f, 0.f, 0.f};
#pragma unroll
    for (int i = 0; i < 16; ++i) {
        float p = re[i] * re[i] + im[i] * im[i];
#pragma unroll
        for (int q = 0; q < 4; ++q) ev[q] += ((i >> (3 - q)) & 1) ? -p : p;
    }
#pragma unroll
    for (int j = 0; j < 4; ++j) {
        float o = fcb[j];
#pragma unroll
        for (int q = 0; q < 4; ++q) o += ev[q] * fcw[j * 4 + q];
        out_pre[b * 4 + j] = o;
        atomicAdd(&stats[48 + j], o);
        atomicAdd(&stats[52 + j], o * o);
    }
}

// ---------------- final batchnorm over batch ------------------------------
__global__ void qn_bn3(const float* __restrict__ out_pre,
                       const float* __restrict__ stats,
                       const float* __restrict__ g3,
                       const float* __restrict__ b3,
                       float* __restrict__ out) {
    int i = blockIdx.x * 256 + threadIdx.x;  // 16384
    int j = i & 3;
    const float invN = 1.f / 4096.f;
    float mean = stats[48 + j] * invN;
    float var  = stats[52 + j] * invN - mean * mean;
    out[i] = (out_pre[i] - mean) * rsqrtf(var + EPSB) * g3[j] + b3[j];
}

extern "C" void kernel_launch(void* const* d_in, const int* in_sizes, int n_in,
                              void* d_out, int out_size, void* d_ws, size_t ws_size,
                              hipStream_t stream) {
    const float* x   = (const float*)d_in[0];
    const float* w1  = (const float*)d_in[1];
    const float* b1  = (const float*)d_in[2];
    const float* g1  = (const float*)d_in[3];
    const float* be1 = (const float*)d_in[4];
    const float* w2  = (const float*)d_in[5];
    const float* b2  = (const float*)d_in[6];
    const float* g2  = (const float*)d_in[7];
    const float* be2 = (const float*)d_in[8];
    const float* rz  = (const float*)d_in[9];
    const float* fcw = (const float*)d_in[10];
    const float* fcb = (const float*)d_in[11];
    const float* g3  = (const float*)d_in[12];
    const float* b3  = (const float*)d_in[13];
    float* out = (float*)d_out;

    char* base = (char*)d_ws;
    float*     stats  = (float*)base;                         // 256 B
    _Float16*  h1     = (_Float16*)(base + 256);              // 67,108,864 B
    _Float16*  h1p    = (_Float16*)(base + 256 + 67108864u);  // 16,777,216 B
    _Float16*  h2     = (_Float16*)(base + 256 + 67108864u + 16777216u);   // 33,554,432 B
    float*     encode = (float*)(base + 256 + 67108864u + 16777216u + 33554432u); // 16 KB
    float*     outpre = (float*)(base + 256 + 67108864u + 16777216u + 33554432u + 16384u);

    qn_init_stats<<<1, 64, 0, stream>>>(stats);
    qn_conv1_wmma<<<65536, 128, 0, stream>>>(x, w1, b1, h1);
    qn_stats1<<<32768, 256, 0, stream>>>(h1, stats);
    qn_pool1<<<32768, 256, 0, stream>>>(h1, stats, g1, be1, h1p);
    qn_conv2_wmma<<<16384, 128, 0, stream>>>(h1p, w2, b2, h2);
    qn_stats2<<<65536, 256, 0, stream>>>(h2, stats);
    qn_encode<<<4096, 256, 0, stream>>>(h2, stats, g2, be2, encode);
    qn_quantum_fc<<<16, 256, 0, stream>>>(encode, rz, fcw, fcb, outpre, stats);
    qn_bn3<<<64, 256, 0, stream>>>(outpre, stats, g3, b3, out);
}
